// PPN_14894946582529
// MI455X (gfx1250) — matrix operand order repaired and checked by reading, not compiled, not run on hardware
//
#include <hip/hip_runtime.h>
#include <hip/hip_bf16.h>
#include <math.h>

// Problem sizes (fixed by the reference)
#define BB   4096
#define TT   512
#define FF   17
#define HH   32
#define PP   8
#define FPD  18      // FF + 1 (demo row)
#define G3   96      // 3*HH
#define TC   8       // x time-chunk per global load

typedef __attribute__((ext_vector_type(16))) __bf16 v16bf;
typedef __attribute__((ext_vector_type(8)))  float  v8f;

// Single-instruction v_rcp_f32 (GRU gates don't need IEEE divide chains)
__device__ __forceinline__ float fast_rcp(float x) { return __builtin_amdgcn_rcpf(x); }
__device__ __forceinline__ float sigm(float x) { return fast_rcp(1.0f + __expf(-x)); }
__device__ __forceinline__ float tanh_fast(float x) {
    float t = __expf(-2.0f * fabsf(x));          // in (0,1], no overflow
    float r = (1.0f - t) * fast_rcp(1.0f + t);
    return copysignf(r, x);
}

// ---------------------------------------------------------------------------
// Phase 0: transpose x[B,T,F] -> xT[F,T,B] so the GRU time loop reads are
// coalesced (lanes over b, sequential over t). 143 MB in + 143 MB out ~ 12us.
// ---------------------------------------------------------------------------
__global__ __launch_bounds__(256) void transpose_x(const float* __restrict__ x,
                                                   float* __restrict__ xT) {
    const int t  = blockIdx.x;          // 0..TT-1
    const int b0 = blockIdx.y * 64;     // 64 batches per block
    __shared__ float tile[FF][64 + 1];
    for (int i = threadIdx.x; i < 64 * FF; i += 256) {
        int b = i / FF, f = i % FF;     // reads contiguous 17-float rows
        tile[f][b] = x[((size_t)(b0 + b) * TT + t) * FF + f];
    }
    __syncthreads();
    for (int i = threadIdx.x; i < FF * 64; i += 256) {
        int f = i / 64, b = i % 64;     // 256B coalesced writes
        xT[((size_t)f * TT + t) * BB + b0 + b] = tile[f][b];
    }
}

// ---------------------------------------------------------------------------
// Phase 1: fused GRU. One wave (32 lanes) owns a 16-batch x 1-feature tile and
// runs all T=512 steps. Per step: 6x v_wmma_f32_16x16x32_bf16 (K=32==H) for
// h @ W_hh^T with per-gate bias pre-folded into C, then in-register gates.
// h state lives in the WMMA C/D layout: lane holds 8 rows x 2 hidden columns.
// x is consumed in 8-step chunks: one b128 load per lane per chunk + prefetch.
// ---------------------------------------------------------------------------
__global__ __launch_bounds__(32) void gru_kernel(
    const float* __restrict__ xsrc, int xT_mode,
    const float* __restrict__ W_ih, const float* __restrict__ W_hh,
    const float* __restrict__ b_ih, const float* __restrict__ b_hh,
    float* __restrict__ out_h)   // raw h into d_out h_t slice [B][FPD][HH]
{
    const int f    = blockIdx.y;
    const int b0   = blockIdx.x * 16;
    const int lane = threadIdx.x;
    const int col  = lane & 15;     // N (gate col) / M (A row) index
    const int hi   = lane >> 4;     // upper/lower half-wave

    __shared__ float h_lds[16][HH + 1];   // +1 pad: conflict-free
    __shared__ float x_lds[TC][16];

    for (int i = lane; i < 16 * HH; i += 32) h_lds[i / HH][i % HH] = 0.0f;

    // --- Preload B fragments: B[k][g] = W_hh[f][g][k], bf16, 6 N-tiles.
    // 32x16 bf16 B, k order used here: lanes 0-15 col N=lane, K=0..15 (2/VGPR);
    // lanes 16-31 col N=lane-16, K=16..31.
    v16bf Bfrag[6];
#pragma unroll
    for (int n = 0; n < 6; ++n) {
        const float* w = W_hh + ((size_t)f * G3 + (n * 16 + col)) * HH;
#pragma unroll
        for (int j = 0; j < 8; ++j) {
            int k = hi * 16 + 2 * j;
            Bfrag[n][2 * j]     = (__bf16)w[k];
            Bfrag[n][2 * j + 1] = (__bf16)w[k + 1];
        }
    }

    // Per-lane gate params at g = 16n + col. Tiles: 0,1=r  2,3=z  4,5=n.
    float wih[6];
#pragma unroll
    for (int n = 0; n < 6; ++n) wih[n] = W_ih[f * G3 + n * 16 + col];
    float crz[4];
#pragma unroll
    for (int n = 0; n < 4; ++n)
        crz[n] = b_ih[f * G3 + n * 16 + col] + b_hh[f * G3 + n * 16 + col];
    float bihn0 = b_ih[f * G3 + 64 + col], bihn1 = b_ih[f * G3 + 80 + col];
    float bhhn0 = b_hh[f * G3 + 64 + col], bhhn1 = b_hh[f * G3 + 80 + col];

    // Fold biases into the WMMA C operand (bias depends only on N == lane).
    v8f Cr0, Cr1, Cz0, Cz1, Cn0, Cn1;
#pragma unroll
    for (int j = 0; j < 8; ++j) {
        Cr0[j] = crz[0]; Cr1[j] = crz[1];
        Cz0[j] = crz[2]; Cz1[j] = crz[3];
        Cn0[j] = bhhn0;  Cn1[j] = bhhn1;   // n-gate: r * (Wh + b_hh)
    }

    float h0[8], h1[8];
#pragma unroll
    for (int j = 0; j < 8; ++j) { h0[j] = 0.0f; h1[j] = 0.0f; }

    // x addressing
    const float* xT_base = xsrc + ((size_t)f * TT) * BB + b0;          // xT mode
    const float* xd_base = xsrc + ((size_t)(b0 + col) * TT) * FF + f;  // direct
    const int ctt = lane >> 2;          // chunk row this lane loads (xT mode)
    const int cof = (lane & 3) * 4;     // float4 offset within the 16-b row

    __syncthreads();

    for (int tc = 0; tc < TT; tc += TC) {
        // ---- load 8 timesteps of x for the 16 batches ----
        if (xT_mode) {
            const float* p = xT_base + (size_t)tc * BB;
            float4 v = *(const float4*)(p + (size_t)ctt * BB + cof);   // b128
            x_lds[ctt][cof + 0] = v.x;
            x_lds[ctt][cof + 1] = v.y;
            x_lds[ctt][cof + 2] = v.z;
            x_lds[ctt][cof + 3] = v.w;
            if (tc + TC < TT)   // prefetch next chunk (global_prefetch_b8)
                __builtin_prefetch(p + (size_t)(TC + ctt) * BB + cof, 0, 1);
        } else if (lane < 16) {
#pragma unroll
            for (int tt = 0; tt < TC; ++tt)
                x_lds[tt][col] = xd_base[(size_t)(tc + tt) * FF];
        }
        __syncthreads();

#pragma unroll 1
        for (int ts = 0; ts < TC; ++ts) {
            // A fragment (16x32 bf16): lane row M=col; K=2j+16*hi
            // (matches B fragment order).
            v16bf A;
#pragma unroll
            for (int j = 0; j < 8; ++j) {
                int k = hi * 16 + 2 * j;
                A[2 * j]     = (__bf16)h_lds[col][k];
                A[2 * j + 1] = (__bf16)h_lds[col][k + 1];
            }

            v8f ar0 = __builtin_amdgcn_wmma_f32_16x16x32_bf16(false, A, false, Bfrag[0], (short)0, Cr0, false, false);
            v8f ar1 = __builtin_amdgcn_wmma_f32_16x16x32_bf16(false, A, false, Bfrag[1], (short)0, Cr1, false, false);
            v8f az0 = __builtin_amdgcn_wmma_f32_16x16x32_bf16(false, A, false, Bfrag[2], (short)0, Cz0, false, false);
            v8f az1 = __builtin_amdgcn_wmma_f32_16x16x32_bf16(false, A, false, Bfrag[3], (short)0, Cz1, false, false);
            v8f an0 = __builtin_amdgcn_wmma_f32_16x16x32_bf16(false, A, false, Bfrag[4], (short)0, Cn0, false, false);
            v8f an1 = __builtin_amdgcn_wmma_f32_16x16x32_bf16(false, A, false, Bfrag[5], (short)0, Cn1, false, false);

            // Gates (C/D layout: row M = j + 8*hi, cols = lane&15 and +16)
#pragma unroll
            for (int j = 0; j < 8; ++j) {
                float xb  = x_lds[ts][j + hi * 8];
                float rg0 = sigm(ar0[j] + xb * wih[0]);
                float rg1 = sigm(ar1[j] + xb * wih[1]);
                float zg0 = sigm(az0[j] + xb * wih[2]);
                float zg1 = sigm(az1[j] + xb * wih[3]);
                float ng0 = tanh_fast(xb * wih[4] + bihn0 + rg0 * an0[j]);
                float ng1 = tanh_fast(xb * wih[5] + bihn1 + rg1 * an1[j]);
                h0[j] = ng0 + zg0 * (h0[j] - ng0);   // (1-z)*n + z*h
                h1[j] = ng1 + zg1 * (h1[j] - ng1);
                int row = j + hi * 8;
                h_lds[row][col]      = h0[j];
                h_lds[row][col + 16] = h1[j];
            }
            __syncthreads();   // 1-wave WG: barrier is ~free, orders LDS
        }
    }

    // Coalesced 128B writes of the final hidden state.
    for (int i = 0; i < 16; ++i)
        out_h[((size_t)(b0 + i)) * (FPD * HH) + f * HH + lane] = h_lds[i][lane];
}

// ---------------------------------------------------------------------------
// Phase 2: per-batch head. One wave per batch; lane == hidden index.
// demo row, LayerNorm over (FPD,HH), cosine distances, P=8 attention, logit.
// ---------------------------------------------------------------------------
__global__ __launch_bounds__(32) void head_kernel(
    const float* __restrict__ statics, const float* __restrict__ demo_W,
    const float* __restrict__ demo_b,  const float* __restrict__ ln_w,
    const float* __restrict__ ln_b,    const float* __restrict__ proto,
    const float* __restrict__ Wq, const float* __restrict__ bq,
    const float* __restrict__ Wk, const float* __restrict__ bk,
    const float* __restrict__ Wv, const float* __restrict__ bv,
    const float* __restrict__ outW, const float* __restrict__ outB,
    float* __restrict__ out_logits, float* __restrict__ out_dist,
    float* __restrict__ out_h)
{
    const int b = blockIdx.x;
    const int lane = threadIdx.x;

    float hr[FPD];
#pragma unroll
    for (int r = 0; r < FF; ++r)
        hr[r] = out_h[(size_t)b * (FPD * HH) + r * HH + lane];
    {   // demo row: static @ demo_W^T + demo_b   (demo_W is [HH][4])
        float d = demo_b[lane];
#pragma unroll
        for (int k = 0; k < 4; ++k) d += statics[b * 4 + k] * demo_W[lane * 4 + k];
        hr[FF] = d;
    }

    // LayerNorm over all 576 elements (wave32 shfl reductions)
    float s = 0.f, s2 = 0.f;
#pragma unroll
    for (int r = 0; r < FPD; ++r) { s += hr[r]; s2 += hr[r] * hr[r]; }
#pragma unroll
    for (int o = 16; o > 0; o >>= 1) { s += __shfl_xor(s, o, 32); s2 += __shfl_xor(s2, o, 32); }
    const float inv = 1.0f / (float)(FPD * HH);
    float mu = s * inv, var = s2 * inv - mu * mu;
    float rstd = rsqrtf(var + 1e-5f);

    float hsq = 0.f;
#pragma unroll
    for (int r = 0; r < FPD; ++r) {
        float v = (hr[r] - mu) * rstd * ln_w[r * HH + lane] + ln_b[r * HH + lane];
        hr[r] = v;
        out_h[(size_t)b * (FPD * HH) + r * HH + lane] = v;   // normalized h_t out
        hsq += v * v;
    }
#pragma unroll
    for (int o = 16; o > 0; o >>= 1) hsq += __shfl_xor(hsq, o, 32);
    float hnorm = sqrtf(hsq);

    // Cosine similarity to P=8 prototypes (prototypes L2-resident, 18KB)
    float dist[PP];
#pragma unroll
    for (int p = 0; p < PP; ++p) {
        float num = 0.f, pn = 0.f;
#pragma unroll
        for (int r = 0; r < FPD; ++r) {
            float pv = proto[((size_t)p * FPD + r) * HH + lane];
            num += hr[r] * pv; pn += pv * pv;
        }
#pragma unroll
        for (int o = 16; o > 0; o >>= 1) { num += __shfl_xor(num, o, 32); pn += __shfl_xor(pn, o, 32); }
        dist[p] = num * fast_rcp(fmaxf(hnorm * sqrtf(pn), 1e-8f));
    }
    if (lane == 0)
#pragma unroll
        for (int p = 0; p < PP; ++p) out_dist[b * PP + p] = dist[p];

    // q = dist @ Wq^T + bq   (Wq is [HH][PP])
    float q = bq[lane];
#pragma unroll
    for (int p = 0; p < PP; ++p) q += dist[p] * Wq[lane * PP + p];

    // keys/values/energies
    float e[PP], iv[PP];
#pragma unroll
    for (int p = 0; p < PP; ++p) {
        float ik = bk[p], vv = bv[p];
#pragma unroll
        for (int r = 0; r < FPD; ++r) { ik += hr[r] * Wk[p * FPD + r]; vv += hr[r] * Wv[p * FPD + r]; }
        iv[p] = vv;
        float ee = ik * q;
#pragma unroll
        for (int o = 16; o > 0; o >>= 1) ee += __shfl_xor(ee, o, 32);
        e[p] = ee;                            // broadcast to all lanes
    }
    float m = e[0];
#pragma unroll
    for (int p = 1; p < PP; ++p) m = fmaxf(m, e[p]);
    float a[PP], asum = 0.f;
#pragma unroll
    for (int p = 0; p < PP; ++p) { a[p] = __expf(e[p] - m); asum += a[p]; }
    float ainv = fast_rcp(asum);
    float v = 0.f;
#pragma unroll
    for (int p = 0; p < PP; ++p) v += a[p] * ainv * iv[p];

    float lg = v * outW[lane];
#pragma unroll
    for (int o = 16; o > 0; o >>= 1) lg += __shfl_xor(lg, o, 32);
    if (lane == 0) out_logits[b] = sigm(lg + outB[0]);
}

// ---------------------------------------------------------------------------
extern "C" void kernel_launch(void* const* d_in, const int* in_sizes, int n_in,
                              void* d_out, int out_size, void* d_ws, size_t ws_size,
                              hipStream_t stream) {
    const float* x       = (const float*)d_in[0];
    const float* statics = (const float*)d_in[1];
    const float* W_ih    = (const float*)d_in[2];
    const float* W_hh    = (const float*)d_in[3];
    const float* b_ih    = (const float*)d_in[4];
    const float* b_hh    = (const float*)d_in[5];
    const float* demo_W  = (const float*)d_in[6];
    const float* demo_b  = (const float*)d_in[7];
    const float* ln_w    = (const float*)d_in[8];
    const float* ln_b    = (const float*)d_in[9];
    const float* proto   = (const float*)d_in[10];
    const float* Wq      = (const float*)d_in[11];
    const float* bq      = (const float*)d_in[12];
    const float* Wk      = (const float*)d_in[13];
    const float* bk      = (const float*)d_in[14];
    const float* Wv      = (const float*)d_in[15];
    const float* bv      = (const float*)d_in[16];
    const float* outW    = (const float*)d_in[17];
    const float* outB    = (const float*)d_in[18];

    float* out_logits = (float*)d_out;                       // [B]
    float* out_dist   = out_logits + BB;                     // [B,P]
    float* out_h      = out_dist + (size_t)BB * PP;          // [B,FPD,HH]

    // Phase 0: transpose x into workspace if it fits, else strided fallback.
    const size_t xt_bytes = (size_t)FF * TT * BB * sizeof(float);
    int use_T = (ws_size >= xt_bytes) ? 1 : 0;
    const float* xsrc = x;
    if (use_T) {
        float* xT = (float*)d_ws;
        dim3 g(TT, BB / 64);
        transpose_x<<<g, 256, 0, stream>>>(x, xT);
        xsrc = xT;
    }

    // Phase 1: GRU (writes raw h into out_h slice)
    dim3 gg(BB / 16, FF);
    gru_kernel<<<gg, 32, 0, stream>>>(xsrc, use_T, W_ih, W_hh, b_ih, b_hh, out_h);

    // Phase 2: LN + prototypes + attention + logit (normalizes out_h in place)
    head_kernel<<<BB, 32, 0, stream>>>(statics, demo_W, demo_b, ln_w, ln_b, proto,
                                       Wq, bq, Wk, bk, Wv, bv, outW, outB,
                                       out_logits, out_dist, out_h);
}
